// OA_71734543777838
// MI455X (gfx1250) — matrix operand order repaired and checked by reading, not compiled, not run on hardware
//
#include <hip/hip_runtime.h>
#include <hip/hip_bf16.h>
#include <cstdint>
#include <cstddef>

typedef _Float16 f16;
typedef __attribute__((ext_vector_type(2)))  _Float16 v2h;
typedef __attribute__((ext_vector_type(8)))  _Float16 v8h;
typedef __attribute__((ext_vector_type(16))) _Float16 v16h;
typedef __attribute__((ext_vector_type(2)))  float    v2f;
typedef __attribute__((ext_vector_type(8)))  float    v8f;

constexpr int BB = 8;
constexpr int CC = 64;
constexpr int NN = 4096;
constexpr int RB = NN / 16;  // 256 row-blocks per batch

union AFrag { v16h v; v8h h[2]; };

static __device__ __forceinline__ v8f wmma_f16(v16h a, v16h b, v8f c) {
  // D = A(16x32 f16) * B(32x16 f16) + C(16x16 f32)
  return __builtin_amdgcn_wmma_f32_16x16x32_f16(false, a, false, b, (short)0, c,
                                                false, false);
}

static __device__ __forceinline__ v8f wmma_f32(v2f a, v2f b, v8f c) {
  // D = A(16x4 f32) * B(4x16 f32) + C(16x16 f32)
  return __builtin_amdgcn_wmma_f32_16x16x4_f32(false, a, false, b, (short)0, c,
                                               false, false);
}

// ---------------------------------------------------------------------------
// K0: one-time prep: Wtrans -> f16, BN constants -> (inv, bias)
// ---------------------------------------------------------------------------
__global__ void k_prep(const float* __restrict__ Wt, const float* __restrict__ gamma,
                       const float* __restrict__ beta, const float* __restrict__ mean,
                       const float* __restrict__ var, f16* __restrict__ Wt16,
                       float* __restrict__ invs, float* __restrict__ biases) {
  int o = threadIdx.x;  // 64 threads
  float inv = gamma[o] * rsqrtf(var[o] + 1e-5f);
  invs[o] = inv;
  biases[o] = beta[o] - mean[o] * inv;
  for (int c = 0; c < CC; ++c) Wt16[o * CC + c] = (f16)Wt[o * CC + c];
}

// ---------------------------------------------------------------------------
// K1: q = Wqk * x (f16, transposed [B][N][C]), v = Wv * x (f32 [B][C][N]).
// Full-f32 GEMM on the matrix pipe: V_WMMA_F32_16X16X4_F32, K=64 -> 16 steps.
// Each wave owns one 16-row o-tile, preloads all W fragments for both weight
// matrices in registers, sweeps 4 n-tiles sharing the x B-fragments.
// ---------------------------------------------------------------------------
__global__ void __launch_bounds__(128) k_qv(const float* __restrict__ x,
                                            const float* __restrict__ Wqk,
                                            const float* __restrict__ Wv,
                                            f16* __restrict__ qT,
                                            float* __restrict__ v) {
  int b = blockIdx.y;
  int nblk = blockIdx.x * 64;
  int tid = threadIdx.x;
  int lane = tid & 31;
  int wave = tid >> 5;
  int r = lane & 15;
  int hi = lane >> 4;
  int o0 = wave * 16;

  // A fragments: f32 A 16x4 layout -> lane holds row M=r, K = {0,1}|{2,3}
  v2f aq[16], av[16];
  const float* wqrow = Wqk + (o0 + r) * CC;
  const float* wvrow = Wv + (o0 + r) * CC;
#pragma unroll
  for (int k = 0; k < 16; ++k) {
    int kk = k * 4 + 2 * hi;
    aq[k] = *(const v2f*)(wqrow + kk);
    av[k] = *(const v2f*)(wvrow + kk);
  }
  const float* xb = x + b * CC * NN;
  for (int nt = 0; nt < 4; ++nt) {
    int n0 = nblk + nt * 16;
    v8f accq = {}, accv = {};
#pragma unroll
    for (int k = 0; k < 16; ++k) {
      int k0 = k * 4 + 2 * hi;  // B 4x16 layout: lane col j=r, K = {0,1}|{2,3}
      v2f bf;
      bf[0] = xb[(k0 + 0) * NN + n0 + r];
      bf[1] = xb[(k0 + 1) * NN + n0 + r];
      accq = wmma_f32(aq[k], bf, accq);
      accv = wmma_f32(av[k], bf, accv);
    }
#pragma unroll
    for (int i = 0; i < 8; ++i) {
      int o = o0 + i + 8 * hi;
      int n = n0 + r;
      qT[((size_t)b * NN + n) * CC + o] = (f16)accq[i];
      v[(size_t)b * CC * NN + o * NN + n] = accv[i];
    }
  }
}

// ---------------------------------------------------------------------------
// K2: fused energy + softmax over a 16-row strip (128 KB strip in LDS).
// ---------------------------------------------------------------------------
__global__ void __launch_bounds__(128) k_energy_softmax(
    const f16* __restrict__ qT, f16* __restrict__ attn,
    float* __restrict__ partial /* [B][RB][NN] */) {
  __shared__ __align__(16) f16 eT[NN][16];  // eT[m][n_local], 128 KB
  __shared__ float pred[128][8];
  __shared__ float rowmax_s[16];
  __shared__ float rrcp_s[16];
  int b = blockIdx.y;
  int n0 = blockIdx.x * 16;
  int tid = threadIdx.x;
  int lane = tid & 31;
  int wave = tid >> 5;
  int r = lane & 15;
  int hi = lane >> 4;
  int off = hi * 16;   // A-fragment K split (contiguous, matches B)
  int koff = hi * 16;  // B-fragment K split (contiguous, matches A)

  // ---- phase 1: WMMA logits -> LDS (transposed, one b128 store per tile)
  const f16* q = qT + (size_t)b * NN * CC;
  const f16* pa = q + (n0 + r) * CC;
  AFrag a0, a1;
  a0.h[0] = *(const v8h*)(pa + off);
  a0.h[1] = *(const v8h*)(pa + 8 + off);
  a1.h[0] = *(const v8h*)(pa + 32 + off);
  a1.h[1] = *(const v8h*)(pa + 40 + off);
  for (int mt = wave * 64; mt < wave * 64 + 64; ++mt) {
    int m0 = mt * 16;
    const f16* pb = q + (m0 + r) * CC;
    v16h b0 = *(const v16h*)(pb + koff);
    v16h b1 = *(const v16h*)(pb + 32 + koff);
    v8f acc = {};
    acc = wmma_f16(a0.v, b0, acc);
    acc = wmma_f16(a1.v, b1, acc);
    v8h hv;
#pragma unroll
    for (int i = 0; i < 8; ++i) hv[i] = (f16)acc[i];
    *(v8h*)(&eT[m0 + r][8 * hi]) = hv;
  }
  __syncthreads();

  // ---- phase 2a: row max, vectorized 8 n-lanes per thread
  int half = tid & 1;   // n range: 8*half .. 8*half+7
  int mrow = tid >> 1;  // 0..63
  {
    v8f pmax;
#pragma unroll
    for (int i = 0; i < 8; ++i) pmax[i] = -3.0e38f;
    for (int j = 0; j < 64; ++j) {
      v8h hv = *(const v8h*)(&eT[mrow + 64 * j][8 * half]);
#pragma unroll
      for (int i = 0; i < 8; ++i) pmax[i] = fmaxf(pmax[i], (float)hv[i]);
    }
#pragma unroll
    for (int i = 0; i < 8; ++i) pred[tid][i] = pmax[i];
  }
  __syncthreads();
  if (tid < 16) {
    int h = tid >> 3, i = tid & 7;
    float mx = pred[h][i];
    for (int s = 1; s < 64; ++s) mx = fmaxf(mx, pred[2 * s + h][i]);
    rowmax_s[tid] = mx;
  }
  __syncthreads();

  // ---- phase 2b: exp in place + row sums, vectorized
  {
    float mx8[8];
#pragma unroll
    for (int i = 0; i < 8; ++i) mx8[i] = rowmax_s[8 * half + i];
    v8f psum = {};
    for (int j = 0; j < 64; ++j) {
      int m = mrow + 64 * j;
      v8h hv = *(const v8h*)(&eT[m][8 * half]);
      v8h ev;
#pragma unroll
      for (int i = 0; i < 8; ++i) {
        float e = __expf((float)hv[i] - mx8[i]);
        ev[i] = (f16)e;
        psum[i] += e;
      }
      *(v8h*)(&eT[m][8 * half]) = ev;
    }
#pragma unroll
    for (int i = 0; i < 8; ++i) pred[tid][i] = psum[i];
  }
  __syncthreads();
  if (tid < 16) {
    int h = tid >> 3, i = tid & 7;
    float s = 0.f;
    for (int p = 0; p < 64; ++p) s += pred[2 * p + h][i];
    rrcp_s[tid] = 1.f / s;
  }
  __syncthreads();

  // ---- phase 3: scale, paired-m writes (b32 stores) + column partials
  float rr[16];
#pragma unroll
  for (int i = 0; i < 16; ++i) rr[i] = rrcp_s[i];
  float* pcol = partial + ((size_t)b * RB + blockIdx.x) * NN;
  for (int mp = tid; mp < NN / 2; mp += 128) {
    int m = mp * 2;
    v16h r0 = *(const v16h*)(&eT[m][0]);
    v16h r1 = *(const v16h*)(&eT[m + 1][0]);
    float cs0 = 0.f, cs1 = 0.f;
#pragma unroll
    for (int i = 0; i < 16; ++i) {
      float v0 = (float)r0[i] * rr[i];
      float v1 = (float)r1[i] * rr[i];
      cs0 += v0;
      cs1 += v1;
      v2h pk;
      pk[0] = (f16)v0;
      pk[1] = (f16)v1;
      *(v2h*)(&attn[((size_t)b * NN + n0 + i) * NN + m]) = pk;
    }
    v2f cs;
    cs[0] = cs0;
    cs[1] = cs1;
    *(v2f*)(&pcol[m]) = cs;
  }
}

// ---------------------------------------------------------------------------
// K3: reduce per-block column partials -> 1/(1e-9 + colsum[m])
// ---------------------------------------------------------------------------
__global__ void k_colsum_reduce(const float* __restrict__ partial,
                                float* __restrict__ crcp) {
  int b = blockIdx.y;
  int m = blockIdx.x * 256 + threadIdx.x;
  float s = 0.f;
  for (int k = 0; k < RB; ++k)
    s += partial[((size_t)b * RB + k) * NN + m];
  crcp[b * NN + m] = 1.f / (1e-9f + s);
}

// ---------------------------------------------------------------------------
// K4: v'[c,m] = f16( v[c,m] / (1e-9 + colsum[m]) )  (folds 2nd normalization)
// ---------------------------------------------------------------------------
__global__ void k_vscale(const float* __restrict__ v, const float* __restrict__ crcp,
                         f16* __restrict__ v16) {
  int idx = blockIdx.x * 256 + threadIdx.x;
  int m = idx & (NN - 1);
  int b = idx / (CC * NN);
  v16[idx] = (f16)(v[idx] * crcp[b * NN + m]);
}

// ---------------------------------------------------------------------------
// K5: y = v' * attn^T (big WMMA), z = Wtrans*y (WMMA via LDS transpose),
//     BN + ReLU + residual, f32 output. One wave per 16-column strip.
// ---------------------------------------------------------------------------
__global__ void k_out(const f16* __restrict__ v16, const f16* __restrict__ attn,
                      const f16* __restrict__ Wt16, const float* __restrict__ invs,
                      const float* __restrict__ biases, const float* __restrict__ xin,
                      float* __restrict__ xout) {
  __shared__ __align__(16) f16 yT[16][80];  // padded: 160B rows, 16B aligned
  int b = blockIdx.y;
  int n0 = blockIdx.x * 16;
  int lane = threadIdx.x;
  int r = lane & 15;
  int hi = lane >> 4;
  int off = hi * 16;
  int koff = hi * 16;
  const f16* vp = v16 + b * CC * NN;
  const f16* ap = attn + (size_t)b * NN * NN + (size_t)(n0 + r) * NN;

  v8f acc[4] = {{}, {}, {}, {}};
  for (int mb = 0; mb < NN; mb += 32) {
    v16h bf = *(const v16h*)(ap + mb + koff);  // attn rows = B columns (n)
#pragma unroll
    for (int cb = 0; cb < 4; ++cb) {
      const f16* p = vp + (cb * 16 + r) * NN + mb;
      AFrag a;
      a.h[0] = *(const v8h*)(p + off);
      a.h[1] = *(const v8h*)(p + 8 + off);
      acc[cb] = wmma_f16(a.v, bf, acc[cb]);
    }
  }
  // stash y tile transposed: yT[n_local][c]
#pragma unroll
  for (int cb = 0; cb < 4; ++cb) {
    v8h hv;
#pragma unroll
    for (int i = 0; i < 8; ++i) hv[i] = (f16)acc[cb][i];
    *(v8h*)(&yT[r][cb * 16 + 8 * hi]) = hv;
  }
  __syncthreads();

  // z^T[n,o] = sum_c yT[n,c] * Wt[o,c]
  AFrag a0, a1;
  const f16* yrow = &yT[r][0];
  a0.h[0] = *(const v8h*)(yrow + off);
  a0.h[1] = *(const v8h*)(yrow + 8 + off);
  a1.h[0] = *(const v8h*)(yrow + 32 + off);
  a1.h[1] = *(const v8h*)(yrow + 40 + off);
#pragma unroll
  for (int ob = 0; ob < 4; ++ob) {
    int o = ob * 16 + r;
    const f16* wrow = Wt16 + o * CC;
    v16h b0 = *(const v16h*)(wrow + koff);
    v16h b1 = *(const v16h*)(wrow + 32 + koff);
    v8f z = {};
    z = wmma_f16(a0.v, b0, z);
    z = wmma_f16(a1.v, b1, z);
    float inv = invs[o];
    float bias = biases[o];
    const float* xi = xin + (b * CC + o) * NN + n0 + 8 * hi;
    float* xo = xout + (b * CC + o) * NN + n0 + 8 * hi;
#pragma unroll
    for (int i = 0; i < 8; ++i)
      xo[i] = fmaxf(z[i] * inv + bias, 0.f) + xi[i];
  }
}

// ---------------------------------------------------------------------------
struct WS {
  f16* qT;
  float* v;
  f16* v16;
  f16* attn;
  float* pcol;
  float* crcp;
  float* out2;
  f16* Wt16;
  float* invs;
  float* biases;
};

static void attention_pass(const float* xin, float* xout, const float* Wqk,
                           const float* Wv, const WS& w, hipStream_t s) {
  k_qv<<<dim3(NN / 64, BB), 128, 0, s>>>(xin, Wqk, Wv, w.qT, w.v);
  k_energy_softmax<<<dim3(RB, BB), 128, 0, s>>>(w.qT, w.attn, w.pcol);
  k_colsum_reduce<<<dim3(NN / 256, BB), 256, 0, s>>>(w.pcol, w.crcp);
  k_vscale<<<(BB * CC * NN) / 256, 256, 0, s>>>(w.v, w.crcp, w.v16);
  k_out<<<dim3(NN / 16, BB), 32, 0, s>>>(w.v16, w.attn, w.Wt16, w.invs,
                                         w.biases, xin, xout);
}

extern "C" void kernel_launch(void* const* d_in, const int* in_sizes, int n_in,
                              void* d_out, int out_size, void* d_ws, size_t ws_size,
                              hipStream_t stream) {
  (void)in_sizes; (void)n_in; (void)out_size; (void)ws_size;
  const float* x     = (const float*)d_in[0];
  const float* Wqk   = (const float*)d_in[1];
  const float* Wv    = (const float*)d_in[2];
  const float* Wt    = (const float*)d_in[3];
  const float* gamma = (const float*)d_in[4];
  const float* beta  = (const float*)d_in[5];
  const float* mean  = (const float*)d_in[6];
  const float* var   = (const float*)d_in[7];

  char* p = (char*)d_ws;
  auto take = [&](size_t bytes) -> char* {
    char* q = p;
    p += (bytes + 255) & ~(size_t)255;
    return q;
  };
  WS w;
  w.qT     = (f16*)take((size_t)BB * NN * CC * 2);
  w.v      = (float*)take((size_t)BB * CC * NN * 4);
  w.v16    = (f16*)take((size_t)BB * CC * NN * 2);
  w.attn   = (f16*)take((size_t)BB * NN * NN * 2);
  w.pcol   = (float*)take((size_t)BB * RB * NN * 4);
  w.crcp   = (float*)take((size_t)BB * NN * 4);
  w.out2   = (float*)take((size_t)BB * CC * NN * 4);
  w.Wt16   = (f16*)take((size_t)CC * CC * 2);
  w.invs   = (float*)take((size_t)CC * 4);
  w.biases = (float*)take((size_t)CC * 4);

  k_prep<<<1, CC, 0, stream>>>(Wt, gamma, beta, mean, var, w.Wt16, w.invs,
                               w.biases);
  // a1 = attn(x); out2 = a1 + x
  attention_pass(x, w.out2, Wqk, Wv, w, stream);
  // a2 = attn(out2); result = a2 + out2
  attention_pass(w.out2, (float*)d_out, Wqk, Wv, w, stream);
}
